// AdaptiveRankLinear_30365418783338
// MI455X (gfx1250) — matrix operand order repaired and checked by reading, not compiled, not run on hardware
//
#include <hip/hip_runtime.h>
#include <hip/hip_bf16.h>

// ---------------------------------------------------------------------------
// AdaptiveRankLinear (TT weight) on MI455X / gfx1250.
// Strategy: materialize W from TT cores (cheap: ~0.6 GFLOP), then one big
// bf16 WMMA GEMM (K extended 3x for bf16-split fp32-accuracy recovery).
// GEMM uses CDNA5 async global->LDS DMA (ASYNCcnt) + double-buffered LDS.
// ---------------------------------------------------------------------------

typedef __attribute__((ext_vector_type(16))) __bf16 v16bf;
typedef __attribute__((ext_vector_type(8)))  float v8f;
typedef int v4i_gcc __attribute__((vector_size(16)));   // matches builtin param

#define KF 4096   // in_features
#define NO 4096   // out_features
#define NBATCH 4096

#ifndef __has_builtin
#define __has_builtin(x) 0
#endif

#if defined(__gfx1250__) && __has_builtin(__builtin_amdgcn_global_load_async_to_lds_b128)
#define USE_ASYNC_COPY 1
#endif

__device__ __forceinline__ unsigned short f32_to_bf16_rne(float f) {
    union { float f; unsigned u; } v; v.f = f;
    unsigned u = v.u + 0x7fffu + ((v.u >> 16) & 1u);
    return (unsigned short)(u >> 16);
}
__device__ __forceinline__ float bf16_to_f32(unsigned short h) {
    union { unsigned u; float f; } v; v.u = ((unsigned)h) << 16;
    return v.f;
}

// Split fp32 -> (hi, lo) bf16 pair:  f ~= hi + lo  (covers ~16 mantissa bits)
__global__ void split_bf16_kernel(const float* __restrict__ in,
                                  unsigned short* __restrict__ hi,
                                  unsigned short* __restrict__ lo, int n) {
    int i = blockIdx.x * blockDim.x + threadIdx.x;
    if (i < n) {
        float f = in[i];
        unsigned short h = f32_to_bf16_rne(f);
        hi[i] = h;
        lo[i] = f32_to_bf16_rne(f - bf16_to_f32(h));
    }
}

// h[i1][n1][i2][n2][s] = sum_r g1[i1,n1,r] * g2[r,i2,n2,s]     (16^5 outputs)
__global__ void tt_h_kernel(const float* __restrict__ g1,
                            const float* __restrict__ g2,
                            float* __restrict__ h) {
    int idx = blockIdx.x * blockDim.x + threadIdx.x;   // < 2^20
    int s  = idx & 15;
    int n2 = (idx >> 4) & 15;
    int i2 = (idx >> 8) & 15;
    int n1 = (idx >> 12) & 15;
    int i1 = (idx >> 16) & 15;
    float acc = 0.f;
#pragma unroll
    for (int r = 0; r < 16; ++r)
        acc += g1[(i1 * 16 + n1) * 16 + r] * g2[((r * 16 + i2) * 16 + n2) * 16 + s];
    h[idx] = acc;
}

// Wt[o][f] = sum_s h[i1,n1,i2,n2,s] * g3[s,i3,n3], stored TRANSPOSED (o-major)
// and pre-split into bf16 hi/lo so the GEMM B-operand reads are contiguous.
__global__ void tt_w_kernel(const float* __restrict__ h,
                            const float* __restrict__ g3,
                            unsigned short* __restrict__ wt_hi,
                            unsigned short* __restrict__ wt_lo) {
    int idx = blockIdx.x * blockDim.x + threadIdx.x;   // o*4096 + f
    int f = idx & 4095;
    int o = idx >> 12;
    int i3 = f & 15, i2 = (f >> 4) & 15, i1 = (f >> 8) & 15;
    int n3 = o & 15, n2 = (o >> 4) & 15, n1 = (o >> 8) & 15;
    const float* hp = &h[(((i1 * 16 + n1) * 16 + i2) * 16 + n2) * 16];
    float acc = 0.f;
#pragma unroll
    for (int s = 0; s < 16; ++s)
        acc += hp[s] * g3[(s * 16 + i3) * 16 + n3];
    unsigned short hb = f32_to_bf16_rne(acc);
    wt_hi[idx] = hb;
    wt_lo[idx] = f32_to_bf16_rne(acc - bf16_to_f32(hb));
}

// ---------------------------------------------------------------------------
// GEMM: out[b][o] = sum_f x[b][f] * Wt[o][f] + bias[o]
// K extended to 3*4096: phase 0: x_hi*W_hi, phase 1: x_hi*W_lo, phase 2: x_lo*W_hi
// Block tile 128x128, BK=32, 256 threads (8 wave32s, 2x4), wave tile 64x32.
// ---------------------------------------------------------------------------
#define BM 128
#define BN 128
#define BK 32
#define LDS_STRIDE 40   // 32 + 8 pad (bf16 elems) -> 80B row pitch
#define KTILES (KF / BK)
#define NT (3 * KTILES)

#ifdef USE_ASYNC_COPY
// Async DMA of one 16B chunk per lane: global -> LDS, tracked by ASYNCcnt.
__device__ __forceinline__ void async_copy16(const void* g, const void* l) {
    __builtin_amdgcn_global_load_async_to_lds_b128(
        (__attribute__((address_space(1))) v4i_gcc*)(uintptr_t)g,
        (__attribute__((address_space(3))) v4i_gcc*)(unsigned int)(uintptr_t)l,
        0, 0);
}
__device__ __forceinline__ void wait_async0() {
#if __has_builtin(__builtin_amdgcn_s_wait_asynccnt)
    __builtin_amdgcn_s_wait_asynccnt(0);
#else
    asm volatile("s_wait_asynccnt 0x0" ::: "memory");
#endif
}
#endif

__global__ __launch_bounds__(256, 2)
void tt_gemm_kernel(const unsigned short* __restrict__ xhi,
                    const unsigned short* __restrict__ xlo,
                    const unsigned short* __restrict__ whi,
                    const unsigned short* __restrict__ wlo,
                    const float* __restrict__ bias,
                    float* __restrict__ out) {
#ifdef USE_ASYNC_COPY
    __shared__ __align__(16) unsigned short As[2][BM * LDS_STRIDE];
    __shared__ __align__(16) unsigned short Bs[2][BN * LDS_STRIDE];
#else
    __shared__ __align__(16) unsigned short As[1][BM * LDS_STRIDE];
    __shared__ __align__(16) unsigned short Bs[1][BN * LDS_STRIDE];
#endif

    const int tid    = threadIdx.x;
    const int lane   = tid & 31;
    const int wave   = tid >> 5;
    const int wm     = wave >> 2;    // 0..1  (M direction, 64 rows each)
    const int wn     = wave & 3;     // 0..3  (N direction, 32 cols each)
    const int lane16 = lane & 15;
    const int half   = lane >> 4;    // lane-half selects K group

    const int blockM = blockIdx.y * BM;   // batch rows
    const int blockN = blockIdx.x * BN;   // output cols

    // per-thread staging coords: 512 16B chunks per tile, 2 per thread
    const int ldRow = tid >> 2;           // 0..63 ; second chunk = +64
    const int ldCol = (tid & 3) * 8;      // bf16 elems

    v8f acc[4][2];
#pragma unroll
    for (int i = 0; i < 4; ++i)
#pragma unroll
        for (int j = 0; j < 2; ++j)
            acc[i][j] = (v8f){0.f, 0.f, 0.f, 0.f, 0.f, 0.f, 0.f, 0.f};

#ifdef USE_ASYNC_COPY
    // ---- CDNA5 path: async global->LDS DMA, double-buffered, 1 barrier/iter
    {
        // preload tile 0 into buffer 0
        const unsigned short* Ap = xhi;
        const unsigned short* Bp = whi;
        async_copy16(&Ap[(size_t)(blockM + ldRow) * KF + ldCol],        &As[0][ldRow * LDS_STRIDE + ldCol]);
        async_copy16(&Ap[(size_t)(blockM + ldRow + 64) * KF + ldCol],   &As[0][(ldRow + 64) * LDS_STRIDE + ldCol]);
        async_copy16(&Bp[(size_t)(blockN + ldRow) * KF + ldCol],        &Bs[0][ldRow * LDS_STRIDE + ldCol]);
        async_copy16(&Bp[(size_t)(blockN + ldRow + 64) * KF + ldCol],   &Bs[0][(ldRow + 64) * LDS_STRIDE + ldCol]);
    }
    for (int kt = 0; kt < NT; ++kt) {
        const int cur = kt & 1;
        wait_async0();          // this wave's DMA for tile kt has landed
        __syncthreads();        // everyone's landed; everyone done reading cur^1

        if (kt + 1 < NT) {      // kick DMA for tile kt+1 into the other buffer
            const int kn    = kt + 1;
            const int phase = kn / KTILES;
            const int k0    = (kn % KTILES) * BK;
            const unsigned short* Ap = (phase < 2) ? xhi : xlo;
            const unsigned short* Bp = (phase == 1) ? wlo : whi;
            const int nxt = cur ^ 1;
            async_copy16(&Ap[(size_t)(blockM + ldRow) * KF + k0 + ldCol],      &As[nxt][ldRow * LDS_STRIDE + ldCol]);
            async_copy16(&Ap[(size_t)(blockM + ldRow + 64) * KF + k0 + ldCol], &As[nxt][(ldRow + 64) * LDS_STRIDE + ldCol]);
            async_copy16(&Bp[(size_t)(blockN + ldRow) * KF + k0 + ldCol],      &Bs[nxt][ldRow * LDS_STRIDE + ldCol]);
            async_copy16(&Bp[(size_t)(blockN + ldRow + 64) * KF + k0 + ldCol], &Bs[nxt][(ldRow + 64) * LDS_STRIDE + ldCol]);
        }

        // --- LDS -> fragments (ISA 7.12.2 layouts) ---
        v16bf afr[4];
#pragma unroll
        for (int i = 0; i < 4; ++i) {
            const int m = wm * 64 + i * 16 + lane16;
            uint4 c0 = *(const uint4*)&As[cur][m * LDS_STRIDE + half * 8];
            uint4 c1 = *(const uint4*)&As[cur][m * LDS_STRIDE + 16 + half * 8];
            ((uint4*)&afr[i])[0] = c0;
            ((uint4*)&afr[i])[1] = c1;
        }
        v16bf bfr[2];
#pragma unroll
        for (int j = 0; j < 2; ++j) {
            const int n = wn * 32 + j * 16 + lane16;
            uint4 c0 = *(const uint4*)&Bs[cur][n * LDS_STRIDE + half * 8];
            uint4 c1 = *(const uint4*)&Bs[cur][n * LDS_STRIDE + 16 + half * 8];
            ((uint4*)&bfr[j])[0] = c0;
            ((uint4*)&bfr[j])[1] = c1;
        }
#pragma unroll
        for (int i = 0; i < 4; ++i)
#pragma unroll
            for (int j = 0; j < 2; ++j)
                acc[i][j] = __builtin_amdgcn_wmma_f32_16x16x32_bf16(
                    false, afr[i], false, bfr[j], (short)0, acc[i][j], false, false);
    }
#else
    // ---- fallback: VGPR-staged copy, single LDS buffer (proven path)
    for (int kt = 0; kt < NT; ++kt) {
        const int phase = kt / KTILES;
        const int k0    = (kt % KTILES) * BK;
        const unsigned short* Aptr = (phase < 2) ? xhi : xlo;
        const unsigned short* Bptr = (phase == 1) ? wlo : whi;

#pragma unroll
        for (int c = 0; c < 2; ++c) {
            const int row = ldRow + c * 64;
            uint4 a = *(const uint4*)&Aptr[(size_t)(blockM + row) * KF + k0 + ldCol];
            *(uint4*)&As[0][row * LDS_STRIDE + ldCol] = a;
            uint4 b = *(const uint4*)&Bptr[(size_t)(blockN + row) * KF + k0 + ldCol];
            *(uint4*)&Bs[0][row * LDS_STRIDE + ldCol] = b;
        }
        __syncthreads();

        v16bf afr[4];
#pragma unroll
        for (int i = 0; i < 4; ++i) {
            const int m = wm * 64 + i * 16 + lane16;
            uint4 c0 = *(const uint4*)&As[0][m * LDS_STRIDE + half * 8];
            uint4 c1 = *(const uint4*)&As[0][m * LDS_STRIDE + 16 + half * 8];
            ((uint4*)&afr[i])[0] = c0;
            ((uint4*)&afr[i])[1] = c1;
        }
        v16bf bfr[2];
#pragma unroll
        for (int j = 0; j < 2; ++j) {
            const int n = wn * 32 + j * 16 + lane16;
            uint4 c0 = *(const uint4*)&Bs[0][n * LDS_STRIDE + half * 8];
            uint4 c1 = *(const uint4*)&Bs[0][n * LDS_STRIDE + 16 + half * 8];
            ((uint4*)&bfr[j])[0] = c0;
            ((uint4*)&bfr[j])[1] = c1;
        }
#pragma unroll
        for (int i = 0; i < 4; ++i)
#pragma unroll
            for (int j = 0; j < 2; ++j)
                acc[i][j] = __builtin_amdgcn_wmma_f32_16x16x32_bf16(
                    false, afr[i], false, bfr[j], (short)0, acc[i][j], false, false);
        __syncthreads();
    }
#endif

    // --- epilogue: C/D layout (ISA): VGPR e -> row e (lanes 0-15) / row 8+e ---
#pragma unroll
    for (int i = 0; i < 4; ++i) {
        const int mBase = blockM + wm * 64 + i * 16 + half * 8;
#pragma unroll
        for (int j = 0; j < 2; ++j) {
            const int col = blockN + wn * 32 + j * 16 + lane16;
            const float bv = bias[col];
#pragma unroll
            for (int e = 0; e < 8; ++e)
                out[(size_t)(mBase + e) * NO + col] = acc[i][j][e] + bv;
        }
    }
}

extern "C" void kernel_launch(void* const* d_in, const int* in_sizes, int n_in,
                              void* d_out, int out_size, void* d_ws, size_t ws_size,
                              hipStream_t stream) {
    (void)in_sizes; (void)n_in; (void)out_size; (void)ws_size;
    const float* x    = (const float*)d_in[0];
    const float* g1   = (const float*)d_in[1];
    const float* g2   = (const float*)d_in[2];
    const float* g3   = (const float*)d_in[3];
    const float* bias = (const float*)d_in[4];

    const size_t MAT = (size_t)4096 * 4096;
    unsigned short* xhi = (unsigned short*)d_ws;          // 32MB
    unsigned short* xlo = xhi + MAT;                      // 32MB
    unsigned short* whi = xlo + MAT;                      // 32MB
    unsigned short* wlo = whi + MAT;                      // 32MB
    float*          hbuf = (float*)(wlo + MAT);           // 4MB (16^5 fp32)

    // 1) split x into bf16 hi/lo
    split_bf16_kernel<<<(int)(MAT / 256), 256, 0, stream>>>(x, xhi, xlo, (int)MAT);
    // 2) h = g1 x g2  (tiny)
    tt_h_kernel<<<(1 << 20) / 256, 256, 0, stream>>>(g1, g2, hbuf);
    // 3) Wt (transposed) = h x g3, split into bf16 hi/lo
    tt_w_kernel<<<(int)(MAT / 256), 256, 0, stream>>>(hbuf, g3, whi, wlo);
    // 4) big WMMA GEMM with bf16x3 accuracy recovery
    dim3 grid(NO / BN, NBATCH / BM);
    tt_gemm_kernel<<<grid, 256, 0, stream>>>(xhi, xlo, whi, wlo, bias, (float*)d_out);
}
